// SRUCell_49907519979724
// MI455X (gfx1250) — matrix operand order repaired and checked by reading, not compiled, not run on hardware
//
#include <hip/hip_runtime.h>
#include <hip/hip_bf16.h>

// ---------------- types ----------------
typedef __attribute__((ext_vector_type(16))) __bf16 bf16x16;
typedef __attribute__((ext_vector_type(8)))  float  f32x8;

union FragAB {
    bf16x16 v;
    uint4   q[2];
};

__device__ __forceinline__ float bf2f(unsigned short s) {
    return __uint_as_float(((unsigned)s) << 16);
}

// fp32x2 -> packed bf16x2, round-half-up: 2x v_add + 1x v_perm_b32
__device__ __forceinline__ unsigned pack2bf(float a, float b) {
    unsigned ua = __float_as_uint(a) + 0x8000u;
    unsigned ub = __float_as_uint(b) + 0x8000u;
    // D[15:0] = ua[31:16], D[31:16] = ub[31:16]
    return __builtin_amdgcn_perm(ub, ua, 0x07060302u);
}

// gfx1250 V_TANH_F32 probe
#if defined(__has_builtin)
#if __has_builtin(__builtin_amdgcn_tanhf)
#define TANHF(x) __builtin_amdgcn_tanhf(x)
#elif __has_builtin(__builtin_amdgcn_tanh_f32)
#define TANHF(x) __builtin_amdgcn_tanh_f32(x)
#endif
#endif

// ---------------- problem dims (fixed by reference) ----------------
static constexpr int Ls = 2048;
static constexpr int Bs = 32;
static constexpr int Ds = 512;
static constexpr int Ms = Ls * Bs;           // 65536
static constexpr int Ns = 3 * Ds;            // 1536
static constexpr int Ks = Ds;                // 512
static constexpr size_t PLANE = (size_t)Ms * Ds;  // 33554432 elems per U plane

// ws layout (ushort units): [0, 3*PLANE) = U planes; [3*PLANE, +Ns*Ks) = W~ (permuted B^T, bf16)
static constexpr size_t WT_OFF = 3 * PLANE;

// ---------------- pre-pass: W~[n~][k] = bf16(W[k][3*(n~&511) + (n~>>9)]) ----------------
__global__ __launch_bounds__(256)
void sru_wprep(const float* __restrict__ Wg, unsigned short* __restrict__ Wt) {
    int t = blockIdx.x * 256 + threadIdx.x;          // 0 .. Ns*Ks/2-1
    int nt = t >> 8;                                 // n~
    int k2 = (t & 255) << 1;                         // k
    int srccol = 3 * (nt & 511) + (nt >> 9);
    float a = Wg[(size_t)k2 * Ns + srccol];
    float b = Wg[(size_t)(k2 + 1) * Ns + srccol];
    *reinterpret_cast<unsigned*>(&Wt[(size_t)nt * Ks + k2]) = pack2bf(a, b);
}

// ---------------- GEMM: U~ = bf16( x @ W_perm ) via v_wmma_f32_16x16x32_bf16 ----------
static constexpr int LDT = 40;   // padded LDS row stride in bf16 units (80B = 20 banks, conflict-free)

__global__ __launch_bounds__(256)
void sru_gemm_wmma(const float* __restrict__ xg, const unsigned short* __restrict__ Wt,
                   unsigned short* __restrict__ U) {
    __shared__ __align__(16) unsigned short lA[128 * LDT];   // A tile: [row 0..127][k 0..31]
    __shared__ __align__(16) unsigned short lB[128 * LDT];   // B^T tile: [col 0..127][k 0..31]

    const int tid  = threadIdx.x;
    const int lane = tid & 31;
    const int wid  = tid >> 5;
    const int l15  = lane & 15;
    const int lh   = lane >> 4;          // 0 or 1

    const int nTile = blockIdx.x;        // 0..11
    const int mTile = blockIdx.y;        // 0..511
    const int m0 = mTile * 128;
    const int n0 = nTile * 128;          // base in n~ space
    const int kplane = n0 >> 9;
    const int d0 = n0 & 511;

    const int waveM = wid >> 1;          // 0..3 -> 32 rows each
    const int waveN = wid & 1;           // 0..1 -> 64 cols each

    // A: 128 rows x 8 float4 => 1024 chunks, 4 per thread
    const int arow0 = tid >> 3, ac4 = tid & 7;        // + i*32 rows
    // B: 128 cols x 4 b128-chunks => 512 chunks, 2 per thread
    const int bcol0 = tid >> 2, bkc = tid & 3;        // + i*64 cols

    f32x8 acc[2][4];
    #pragma unroll
    for (int i = 0; i < 2; ++i)
        #pragma unroll
        for (int j = 0; j < 4; ++j) acc[i][j] = (f32x8)(0.0f);

    float4 ra[4];
    uint4  rb[2];

    auto loadG = [&](int k0) {
        #pragma unroll
        for (int i = 0; i < 4; ++i)
            ra[i] = *reinterpret_cast<const float4*>(
                xg + (size_t)(m0 + arow0 + i * 32) * Ks + k0 + ac4 * 4);
        #pragma unroll
        for (int i = 0; i < 2; ++i)
            rb[i] = *reinterpret_cast<const uint4*>(
                Wt + (size_t)(n0 + bcol0 + i * 64) * Ks + k0 + bkc * 8);
    };

    loadG(0);

    for (int kk = 0; kk < Ks / 32; ++kk) {
        __syncthreads();                 // previous iter's LDS reads done
        // ---- store staged tile to LDS ----
        #pragma unroll
        for (int i = 0; i < 4; ++i) {
            uint2 p;
            p.x = pack2bf(ra[i].x, ra[i].y);
            p.y = pack2bf(ra[i].z, ra[i].w);
            *reinterpret_cast<uint2*>(&lA[(arow0 + i * 32) * LDT + ac4 * 4]) = p;
        }
        #pragma unroll
        for (int i = 0; i < 2; ++i)
            *reinterpret_cast<uint4*>(&lB[(bcol0 + i * 64) * LDT + bkc * 8]) = rb[i];
        __syncthreads();

        if (kk + 1 < Ks / 32) loadG((kk + 1) * 32);   // prefetch next tile into regs

        // ---- load fragments ----
        FragAB a[2], b[4];
        #pragma unroll
        for (int sm = 0; sm < 2; ++sm) {
            int row = waveM * 32 + sm * 16 + l15;
            const uint4* p = reinterpret_cast<const uint4*>(&lA[row * LDT + lh * 8]);
            a[sm].q[0] = p[0];   // K = 8*lh + 0..7
            a[sm].q[1] = p[2];   // K = 16 + 8*lh + 0..7   (+32 bytes)
        }
        #pragma unroll
        for (int sn = 0; sn < 4; ++sn) {
            int col = waveN * 64 + sn * 16 + l15;
            const uint4* p = reinterpret_cast<const uint4*>(&lB[col * LDT + lh * 8]);
            b[sn].q[0] = p[0];   // K = 8*lh + 0..7
            b[sn].q[1] = p[2];   // K = 16 + 8*lh + 0..7   (+32 bytes)
        }
        // ---- 8 WMMAs ----
        #pragma unroll
        for (int sm = 0; sm < 2; ++sm)
            #pragma unroll
            for (int sn = 0; sn < 4; ++sn)
                acc[sm][sn] = __builtin_amdgcn_wmma_f32_16x16x32_bf16(
                    false, a[sm].v, false, b[sn].v, (short)0, acc[sm][sn], false, false);
    }

    // ---- store: plane kplane, bf16; D layout: VGPR r -> M = r + 8*lh, N = l15 ----
    unsigned short* Up = U + (size_t)kplane * PLANE;
    #pragma unroll
    for (int sm = 0; sm < 2; ++sm) {
        int rowb = m0 + waveM * 32 + sm * 16 + lh * 8;
        #pragma unroll
        for (int sn = 0; sn < 4; ++sn) {
            int colb = d0 + waveN * 64 + sn * 16 + l15;
            #pragma unroll
            for (int r = 0; r < 8; r += 2) {
                unsigned pk = pack2bf(acc[sm][sn][r], acc[sm][sn][r + 1]);
                Up[(size_t)(rowb + r) * Ds + colb]     = (unsigned short)pk;
                Up[(size_t)(rowb + r + 1) * Ds + colb] = (unsigned short)(pk >> 16);
            }
        }
    }
}

// ---------------- Scan: per-(b,d) recurrence over L, ping-pong pipelined ----------------
// Critical path per step (tanh path): v_fma -> v_tanh_f32 -> v_fma
static constexpr int CH = 16;

__global__ __launch_bounds__(256)
void sru_scan(const float* __restrict__ xg, const float* __restrict__ wc,
              const float* __restrict__ bias, const unsigned short* __restrict__ U,
              float* __restrict__ out) {
    const int tid = blockIdx.x * 256 + threadIdx.x;   // 0..16383 == b*512 + d
    const int d = tid & (Ds - 1);

#ifdef TANHF
    const float wcf_c = 0.5f * wc[d],  wcr_c = 0.5f * wc[Ds + d];   // half-arg gains
    const float bf_c  = 0.5f * bias[d], br_c = 0.5f * bias[Ds + d];
    const float ksc   = 0.5f;                                       // u pre-scale in load
#else
    const float L2E = 1.4426950408889634f;
    const float wcf_c = -L2E * wc[d],  wcr_c = -L2E * wc[Ds + d];   // exp2-domain gains
    const float bf_c  = -L2E * bias[d], br_c = -L2E * bias[Ds + d];
    const float ksc   = -L2E;
#endif

    const unsigned short* U0 = U;
    const unsigned short* U1 = U + PLANE;
    const unsigned short* U2 = U + 2 * PLANE;
    float* hout = out;
    float* cout = out + PLANE;

    const int STRIDE = Bs * Ds;   // 16384 elements per timestep
    const int NT = Ls / CH;       // 128 chunks

    float c = 0.0f;

    float A0[CH], A1[CH], A2[CH], AX[CH];
    float B0[CH], B1[CH], B2[CH], BX[CH];

    // load phase also pre-transforms gate biases into the transcendental's domain
    auto loadCh = [&](int chunk, float* a0, float* a1, float* a2, float* ax) {
        #pragma unroll
        for (int j = 0; j < CH; ++j) {
            size_t o = ((size_t)chunk * CH + j) * STRIDE + tid;
            a0[j] = bf2f(__builtin_nontemporal_load(&U0[o]));
            a1[j] = ksc * bf2f(__builtin_nontemporal_load(&U1[o])) + bf_c;
            a2[j] = ksc * bf2f(__builtin_nontemporal_load(&U2[o])) + br_c;
            ax[j] = __builtin_nontemporal_load(&xg[o]);
        }
    };
    auto compCh = [&](int chunk, const float* a0, const float* a1, const float* a2,
                      const float* ax) {
        #pragma unroll
        for (int j = 0; j < CH; ++j) {
#ifdef TANHF
            float zf = wcf_c * c + a1[j];            // 0.5*(u1 + wcf*c + bf)
            float zr = wcr_c * c + a2[j];
            float tf = TANHF(zf);
            float tr = TANHF(zr);                    // independent of tf chain
            float hs = 0.5f * (c - a0[j]);           // overlaps tanh latency
            float hm = 0.5f * (c + a0[j]);
            c = tf * hs + hm;                        // fg*c + (1-fg)*u0
            float h = tr * (0.5f * (c - ax[j])) + 0.5f * (c + ax[j]);
#else
            float ef = __builtin_amdgcn_exp2f(wcf_c * c + a1[j]);
            float er = __builtin_amdgcn_exp2f(wcr_c * c + a2[j]);
            float fg = __builtin_amdgcn_rcpf(1.0f + ef);
            float rg = __builtin_amdgcn_rcpf(1.0f + er);
            c = fg * (c - a0[j]) + a0[j];
            float h = rg * (c - ax[j]) + ax[j];
#endif
            __builtin_nontemporal_store(h, &hout[((size_t)chunk * CH + j) * STRIDE + tid]);
        }
    };

    loadCh(0, A0, A1, A2, AX);
    for (int t = 0; t < NT; t += 2) {
        int t1 = (t + 1 < NT) ? t + 1 : t;     // clamped re-read on the tail (harmless)
        int t2 = (t + 2 < NT) ? t + 2 : t;
        loadCh(t1, B0, B1, B2, BX);            // prefetch chunk t+1
        compCh(t, A0, A1, A2, AX);             // compute chunk t
        loadCh(t2, A0, A1, A2, AX);            // prefetch chunk t+2
        compCh(t + 1, B0, B1, B2, BX);         // compute chunk t+1
    }
    cout[tid] = c;   // c_last (B, D)
}

// ---------------- launcher ----------------
extern "C" void kernel_launch(void* const* d_in, const int* in_sizes, int n_in,
                              void* d_out, int out_size, void* d_ws, size_t ws_size,
                              hipStream_t stream) {
    const float* x    = (const float*)d_in[0];
    const float* W    = (const float*)d_in[1];
    const float* wc   = (const float*)d_in[2];
    const float* bias = (const float*)d_in[3];
    float* out = (float*)d_out;
    unsigned short* U  = (unsigned short*)d_ws;      // 3 planes bf16 = 192 MiB
    unsigned short* Wt = U + WT_OFF;                 // permuted B^T bf16 = 1.5 MiB

    sru_wprep<<<(Ns * Ks / 2) / 256, 256, 0, stream>>>(W, Wt);

    dim3 gGemm(Ns / 128, Ms / 128);   // (12, 512): nTile fastest -> x tiles reuse in L2
    sru_gemm_wmma<<<gGemm, 256, 0, stream>>>(x, Wt, U);

    dim3 gScan((Bs * Ds) / 256);      // 64 blocks * 256 threads = 16384 channels
    sru_scan<<<gScan, 256, 0, stream>>>(x, wc, bias, U, out);
}
